// BertSelfAttention_89197880803839
// MI455X (gfx1250) — compile-verified, matrix operand order checked by reading, not run on hardware
//
#include <hip/hip_runtime.h>
#include <hip/hip_bf16.h>
#include <math.h>

// ---------------------------------------------------------------------------
// BERT self-attention (single head), B=4, S=4096, D=512, fp32 in/out.
// f16 WMMA (v_wmma_f32_16x16x32_f16) with fp32 accumulate, flash-attention.
// ---------------------------------------------------------------------------

typedef __attribute__((ext_vector_type(16))) _Float16 v16h;
typedef __attribute__((ext_vector_type(8)))  _Float16 v8h;
typedef __attribute__((ext_vector_type(8)))  float    v8f;
typedef __attribute__((ext_vector_type(4)))  _Float16 v4h;

#define BATCH 4
#define SLEN  4096
#define DIM   512
#define SCALE 0.044194173824159216f  // 1/sqrt(512)

// ---- fragment load: 16x32 f16 tile from row-major matrix (A and B share
// the same lane layout: lane&15 = row, lane>>4 selects K groups {0..7,16..23}
// vs {8..15,24..31}) ----
__device__ __forceinline__ v16h frag_ld(const _Float16* __restrict__ p, int ld, int lane) {
    const int m  = lane & 15;
    const int hs = lane >> 4;
    const _Float16* row = p + (size_t)m * ld + hs * 8;
    v8h lo = *(const v8h*)(row);
    v8h hi = *(const v8h*)(row + 16);
    return __builtin_shufflevector(lo, hi, 0,1,2,3,4,5,6,7,8,9,10,11,12,13,14,15);
}

__device__ __forceinline__ v8f wmma16(v16h a, v16h b, v8f c) {
    return __builtin_amdgcn_wmma_f32_16x16x32_f16(false, a, false, b, (short)0, c, false, false);
}

// ---- fp32 -> f16 conversion, 4 elements/thread ----
__global__ void cvt4_f32_to_f16(const float* __restrict__ in, _Float16* __restrict__ out, int n4) {
    int i = blockIdx.x * blockDim.x + threadIdx.x;
    if (i < n4) {
        float4 v = ((const float4*)in)[i];
        v4h o = { (_Float16)v.x, (_Float16)v.y, (_Float16)v.z, (_Float16)v.w };
        ((v4h*)out)[i] = o;
    }
}

// ---- QKV projection: out[n,e] = sum_d x[n,d]*W[e,d] + b[e] ----
// One 16x16 output tile per wave, 8 waves per block.
// vtmode=0: row-major f16 out [16384,512]; vtmode=1: transposed out [B][D][S].
__global__ __launch_bounds__(256) void qkv_gemm(const _Float16* __restrict__ X,
                                                const _Float16* __restrict__ W,
                                                const float* __restrict__ bias,
                                                _Float16* __restrict__ out,
                                                int vtmode) {
    const int lane = threadIdx.x & 31;
    const int wave = threadIdx.x >> 5;
    const int tile = blockIdx.x * 8 + wave;     // 32768 tiles total
    const int mt = tile >> 5;                   // 1024 row tiles (16 tokens each)
    const int nt = tile & 31;                   // 32 col tiles (16 outputs each)
    const int m16 = lane & 15, hs = lane >> 4;

    const _Float16* Xa = X + (size_t)(mt * 16) * DIM;
    const _Float16* Wb = W + (size_t)(nt * 16) * DIM;

    v8f c = {};
#pragma unroll
    for (int kc = 0; kc < 16; ++kc) {
        v16h a = frag_ld(Xa + kc * 32, DIM, lane);
        v16h b = frag_ld(Wb + kc * 32, DIM, lane);
        c = wmma16(a, b, c);
    }
    const float bv = bias[nt * 16 + m16];       // accumulator column N = lane&15

    if (vtmode == 0) {
        _Float16* o = out + (size_t)(mt * 16) * DIM + nt * 16;
#pragma unroll
        for (int r = 0; r < 8; ++r)
            o[(size_t)(r + 8 * hs) * DIM + m16] = (_Float16)(c[r] + bv);
    } else {
        // V transposed: Vt[b][d][s]
#pragma unroll
        for (int r = 0; r < 8; ++r) {
            int row = mt * 16 + r + 8 * hs;     // global token index
            int b   = row >> 12;                // /4096
            int si  = row & 4095;
            out[((size_t)b * DIM + nt * 16 + m16) * SLEN + si] = (_Float16)(c[r] + bv);
        }
    }
}

// ---- Flash attention: one block (8 waves) per (batch, 16-query tile) ----
__global__ __launch_bounds__(256) void flash_attn(const _Float16* __restrict__ Q,
                                                  const _Float16* __restrict__ K,
                                                  const _Float16* __restrict__ Vt,
                                                  float* __restrict__ out) {
    const int tid  = threadIdx.x;
    const int wave = tid >> 5;
    const int lane = tid & 31;
    const int m16 = lane & 15, hs = lane >> 4;

    const int b  = blockIdx.x >> 8;             // 256 query tiles per batch
    const int qt = blockIdx.x & 255;

    __shared__ float    Sbuf[16][132];          // scaled scores (f32)
    __shared__ _Float16 Pbuf[16][136];          // probabilities (f16), padded
    __shared__ float    m_run[16], l_run[16], corr[16];
    __shared__ float    redbuf[16][16];

    const _Float16* Qb = Q  + ((size_t)b * SLEN + qt * 16) * DIM;
    const _Float16* Kb = K  + (size_t)b * SLEN * DIM;
    const _Float16* Vb = Vt + (size_t)b * DIM * SLEN;

    // cache the full 16x512 Q tile as 16 A-fragments (128 VGPRs)
    v16h qf[16];
#pragma unroll
    for (int c = 0; c < 16; ++c) qf[c] = frag_ld(Qb + c * 32, DIM, lane);

    if (tid < 16) { m_run[tid] = -1e30f; l_run[tid] = 0.0f; }
    v8f acc0 = {}, acc1 = {}, acc2 = {}, acc3 = {};
    __syncthreads();

    for (int kb = 0; kb < SLEN / 128; ++kb) {
        // --- scores: wave w computes S[0:16][w*16 : w*16+16] ---
        v8f s = {};
        const _Float16* Kt = Kb + (size_t)(kb * 128 + wave * 16) * DIM;
#pragma unroll
        for (int c = 0; c < 16; ++c) {
            v16h kf = frag_ld(Kt + c * 32, DIM, lane);
            s = wmma16(qf[c], kf, s);
        }
#pragma unroll
        for (int r = 0; r < 8; ++r)
            Sbuf[r + 8 * hs][wave * 16 + m16] = s[r] * SCALE;
        __syncthreads();

        // --- partial row max (128 threads, 8 segments of 16 per row) ---
        if (tid < 128) {
            int row = tid >> 3, seg = tid & 7;
            float mx = -1e30f;
#pragma unroll
            for (int j = 0; j < 16; ++j) mx = fmaxf(mx, Sbuf[row][seg * 16 + j]);
            redbuf[row][seg] = mx;
        }
        __syncthreads();

        // --- online-softmax state update (16 threads) ---
        if (tid < 16) {
            float tmax = redbuf[tid][0];
#pragma unroll
            for (int j = 1; j < 8; ++j) tmax = fmaxf(tmax, redbuf[tid][j]);
            float mo = m_run[tid];
            float mn = fmaxf(mo, tmax);
            corr[tid]  = __expf(mo - mn);
            m_run[tid] = mn;
        }
        __syncthreads();

        // --- P = exp(S - m), f16 into LDS; partial row sums (256 threads) ---
        {
            int row = tid >> 4, seg = tid & 15;
            float mn = m_run[row];
            float sum = 0.0f;
#pragma unroll
            for (int j = 0; j < 8; ++j) {
                float p = __expf(Sbuf[row][seg * 8 + j] - mn);
                Pbuf[row][seg * 8 + j] = (_Float16)p;
                sum += p;
            }
            redbuf[row][seg] = sum;
        }
        __syncthreads();

        if (tid < 16) {
            float sm = 0.0f;
#pragma unroll
            for (int j = 0; j < 16; ++j) sm += redbuf[tid][j];
            l_run[tid] = l_run[tid] * corr[tid] + sm;
        }
        __syncthreads();

        // --- rescale accumulators, then O += P @ V for this wave's 64 cols ---
        float cr[8];
#pragma unroll
        for (int r = 0; r < 8; ++r) cr[r] = corr[r + 8 * hs];
#pragma unroll
        for (int r = 0; r < 8; ++r) {
            acc0[r] *= cr[r]; acc1[r] *= cr[r]; acc2[r] *= cr[r]; acc3[r] *= cr[r];
        }
#pragma unroll
        for (int kc = 0; kc < 4; ++kc) {
            v16h pf = frag_ld(&Pbuf[0][kc * 32], 136, lane);
            const _Float16* vp = Vb + (size_t)(wave * 64) * SLEN + kb * 128 + kc * 32;
            v16h v0 = frag_ld(vp + (size_t)0  * SLEN, SLEN, lane);
            v16h v1 = frag_ld(vp + (size_t)16 * SLEN, SLEN, lane);
            v16h v2 = frag_ld(vp + (size_t)32 * SLEN, SLEN, lane);
            v16h v3 = frag_ld(vp + (size_t)48 * SLEN, SLEN, lane);
            acc0 = wmma16(pf, v0, acc0);
            acc1 = wmma16(pf, v1, acc1);
            acc2 = wmma16(pf, v2, acc2);
            acc3 = wmma16(pf, v3, acc3);
        }
        __syncthreads();
    }

    // --- finalize: O /= l, write fp32 output ---
    float il[8];
#pragma unroll
    for (int r = 0; r < 8; ++r) il[r] = 1.0f / l_run[r + 8 * hs];
    float* ob = out + ((size_t)b * SLEN + qt * 16) * DIM + wave * 64;
#pragma unroll
    for (int r = 0; r < 8; ++r) {
        int row = r + 8 * hs;
        ob[(size_t)row * DIM + 0 * 16 + m16] = acc0[r] * il[r];
        ob[(size_t)row * DIM + 1 * 16 + m16] = acc1[r] * il[r];
        ob[(size_t)row * DIM + 2 * 16 + m16] = acc2[r] * il[r];
        ob[(size_t)row * DIM + 3 * 16 + m16] = acc3[r] * il[r];
    }
}

// ---------------------------------------------------------------------------
// Workspace layout (f16 buffers), total ~65.5 MiB:
//   Xh  [16384,512]       @ 0
//   Wqh/Wkh/Wvh [512,512] @ 16 MiB
//   Qh  [16384,512]       @ 17.5 MiB
//   Kh  [16384,512]       @ 33.5 MiB
//   Vth [4][512][4096]    @ 49.5 MiB
// ---------------------------------------------------------------------------
static const size_t XH_OFF = 0;
static const size_t WQ_OFF = 16777216;
static const size_t WK_OFF = WQ_OFF + 524288;
static const size_t WV_OFF = WK_OFF + 524288;
static const size_t Q_OFF  = WV_OFF + 524288;
static const size_t K_OFF  = Q_OFF + 16777216;
static const size_t VT_OFF = K_OFF + 16777216;

extern "C" void kernel_launch(void* const* d_in, const int* in_sizes, int n_in,
                              void* d_out, int out_size, void* d_ws, size_t ws_size,
                              hipStream_t stream) {
    const float* x  = (const float*)d_in[0];
    const float* Wq = (const float*)d_in[1];
    const float* bq = (const float*)d_in[2];
    const float* Wk = (const float*)d_in[3];
    const float* bk = (const float*)d_in[4];
    const float* Wv = (const float*)d_in[5];
    const float* bv = (const float*)d_in[6];
    float* out = (float*)d_out;

    char* ws = (char*)d_ws;
    _Float16* Xh  = (_Float16*)(ws + XH_OFF);
    _Float16* Wqh = (_Float16*)(ws + WQ_OFF);
    _Float16* Wkh = (_Float16*)(ws + WK_OFF);
    _Float16* Wvh = (_Float16*)(ws + WV_OFF);
    _Float16* Qh  = (_Float16*)(ws + Q_OFF);
    _Float16* Kh  = (_Float16*)(ws + K_OFF);
    _Float16* Vth = (_Float16*)(ws + VT_OFF);

    // 1) convert inputs to f16
    {
        int n4 = (BATCH * SLEN * DIM) / 4;  // 2097152
        cvt4_f32_to_f16<<<n4 / 256, 256, 0, stream>>>(x, Xh, n4);
        int w4 = (DIM * DIM) / 4;           // 65536
        cvt4_f32_to_f16<<<w4 / 256, 256, 0, stream>>>(Wq, Wqh, w4);
        cvt4_f32_to_f16<<<w4 / 256, 256, 0, stream>>>(Wk, Wkh, w4);
        cvt4_f32_to_f16<<<w4 / 256, 256, 0, stream>>>(Wv, Wvh, w4);
    }

    // 2) QKV projections (WMMA f16): 32768 16x16 tiles, 8 per block
    {
        dim3 grid(4096), block(256);
        qkv_gemm<<<grid, block, 0, stream>>>(Xh, Wqh, bq, Qh,  0);
        qkv_gemm<<<grid, block, 0, stream>>>(Xh, Wkh, bk, Kh,  0);
        qkv_gemm<<<grid, block, 0, stream>>>(Xh, Wvh, bv, Vth, 1);
    }

    // 3) flash attention: one block per (batch, 16-query tile)
    {
        dim3 grid(BATCH * (SLEN / 16)), block(256);
        flash_attn<<<grid, block, 0, stream>>>(Qh, Kh, Vth, out);
    }
}
